// attn_90280212561996
// MI455X (gfx1250) — matrix operand order repaired and checked by reading, not compiled, not run on hardware
//
#include <hip/hip_runtime.h>

typedef __attribute__((ext_vector_type(16))) __bf16 v16bf;
typedef __attribute__((ext_vector_type(8)))  __bf16 v8bf;
typedef __attribute__((ext_vector_type(8)))  float  v8f;
typedef __attribute__((ext_vector_type(4)))  float  v4f;
typedef __attribute__((ext_vector_type(4)))  int    v4i;

typedef __attribute__((address_space(3))) v4i lds_v4i;

#if defined(__has_builtin)
#if __has_builtin(__builtin_amdgcn_global_load_async_to_lds_b128)
#define USE_ASYNC_LDS 1
#endif
#endif

// 16-byte global -> LDS copy; async (ASYNCcnt) when available.
static __device__ __forceinline__ void cp16_g2l(void* lds_dst, const void* gsrc) {
#ifdef USE_ASYNC_LDS
    __builtin_amdgcn_global_load_async_to_lds_b128(
        (v4i*)gsrc, (lds_v4i*)lds_dst, 0, 0);
#else
    *(v8bf*)lds_dst = *(const v8bf*)gsrc;
#endif
}
static __device__ __forceinline__ void cp_wait() {
#ifdef USE_ASYNC_LDS
#if __has_builtin(__builtin_amdgcn_s_wait_asynccnt)
    __builtin_amdgcn_s_wait_asynccnt(0);
#else
    asm volatile("s_wait_asynccnt 0x0" ::: "memory");
#endif
#endif
}

static __device__ __forceinline__ v16bf combine16(v8bf lo, v8bf hi) {
    v16bf r;
#pragma unroll
    for (int i = 0; i < 8; ++i) { r[i] = lo[i]; r[i + 8] = hi[i]; }
    return r;
}
// A fragment (16x32 bf16): lane = row (lane&15); elems = 16 contiguous k at hi*16
// -> two 16B chunks at p0, p0+8.
static __device__ __forceinline__ v16bf load_a16(const __bf16* p0) {
    return combine16(*(const v8bf*)p0, *(const v8bf*)(p0 + 8));
}
// B fragment (32x16 bf16): lane = col (lane&15); elems = 16 contiguous k at hi*16.
static __device__ __forceinline__ v16bf load_b16(const __bf16* p) {
    return combine16(*(const v8bf*)p, *(const v8bf*)(p + 8));
}
static __device__ __forceinline__ v8f wmma_bf16(v16bf a, v16bf b, v8f c) {
    return __builtin_amdgcn_wmma_f32_16x16x32_bf16(false, a, false, b,
                                                   (short)0, c, false, false);
}

// ---------------------------------------------------------------------------
// f32 -> bf16 bulk convert (for weights), 8 elems/thread.
// ---------------------------------------------------------------------------
__global__ __launch_bounds__(256) void cvt_bf16_k(const float* __restrict__ src,
                                                  __bf16* __restrict__ dst)
{
    const int i = (blockIdx.x * 256 + threadIdx.x) * 8;
    v4f a = *(const v4f*)(src + i);
    v4f b = *(const v4f*)(src + i + 4);
    v8bf o;
#pragma unroll
    for (int j = 0; j < 4; ++j) { o[j] = (__bf16)a[j]; o[j + 4] = (__bf16)b[j]; }
    *(v8bf*)(dst + i) = o;
}

// ---------------------------------------------------------------------------
// LayerNorm over f=256, wave per row, f32 in -> bf16 out.
// ---------------------------------------------------------------------------
__global__ __launch_bounds__(256) void ln_bf16_k(
    const float* __restrict__ X, const float* __restrict__ g,
    const float* __restrict__ b, __bf16* __restrict__ out, int rows)
{
    const int wave = threadIdx.x >> 5, lane = threadIdx.x & 31;
    const int row = blockIdx.x * 8 + wave;
    if (row >= rows) return;

    const float* xp = X + ((size_t)row << 8) + lane * 8;
    float v[8];
    v4f a0 = *(const v4f*)xp;
    v4f a1 = *(const v4f*)(xp + 4);
#pragma unroll
    for (int i = 0; i < 4; ++i) { v[i] = a0[i]; v[i + 4] = a1[i]; }

    float s = 0.f, s2 = 0.f;
#pragma unroll
    for (int i = 0; i < 8; ++i) { s += v[i]; s2 += v[i] * v[i]; }
#pragma unroll
    for (int m = 1; m < 32; m <<= 1) {
        s  += __shfl_xor(s, m, 32);
        s2 += __shfl_xor(s2, m, 32);
    }
    const float mean = s * (1.f / 256.f);
    const float var  = s2 * (1.f / 256.f) - mean * mean;
    const float rstd = rsqrtf(var + 1e-6f);

    v8bf o;
#pragma unroll
    for (int i = 0; i < 8; ++i) {
        const int c = lane * 8 + i;
        o[i] = (__bf16)((v[i] - mean) * rstd * g[c] + b[c]);
    }
    *(v8bf*)(out + ((size_t)row << 8) + lane * 8) = o;
}

// ---------------------------------------------------------------------------
// GEMM: out[M,256] = A[M,256](bf16) @ Wb[256,256]^T(bf16) + bias.
// 128x64 tile, 8 waves, wave = 16(M)x64(N). W staged to LDS with async copies,
// K-chunk = 64 (8 WMMAs per barrier round). Epilogue fully compile-time.
// ---------------------------------------------------------------------------
template<bool RELU, bool RBROAD, bool HAS_RES, bool OUTB, bool OUTT, bool OUTF>
__global__ __launch_bounds__(256) void gemm_bf16_k(
    const __bf16* __restrict__ A, const __bf16* __restrict__ Wb,
    const float* __restrict__ bias, const float* __restrict__ resid,
    __bf16* __restrict__ outB, __bf16* __restrict__ outT,
    float* __restrict__ outF, int M)
{
    __shared__ __bf16 Wlds[64][72];   // 64 n-rows x 64 k (+8 pad), 144B stride

    const int tid  = threadIdx.x;
    const int wave = tid >> 5, lane = tid & 31;
    const int lo16 = lane & 15, hi = lane >> 4;
    const int mblk = blockIdx.x * 128;
    const int nblk = blockIdx.y * 64;
    const __bf16* Arow = A + ((size_t)(mblk + wave * 16 + lo16) << 8);

    v8f acc[4];
#pragma unroll
    for (int t = 0; t < 4; ++t) acc[t] = v8f{};

    const int nl = tid >> 2;            // W stage: thread -> (n-row, 2x16B)
    const int ks = (tid & 3) * 16;

    for (int k0 = 0; k0 < 256; k0 += 64) {
        const __bf16* wp = Wb + ((size_t)(nblk + nl) << 8) + k0 + ks;
        cp16_g2l(&Wlds[nl][ks],     wp);
        cp16_g2l(&Wlds[nl][ks + 8], wp + 8);
        __builtin_prefetch(Arow + k0 + 64, 0, 1);   // next A chunk
        cp_wait();
        __syncthreads();

#pragma unroll
        for (int kk = 0; kk < 2; ++kk) {
            v16bf aF = load_a16(Arow + k0 + kk * 32 + hi * 16);
#pragma unroll
            for (int t = 0; t < 4; ++t) {
                v16bf bF = load_b16(&Wlds[t * 16 + lo16][kk * 32 + hi * 16]);
                acc[t] = wmma_bf16(aF, bF, acc[t]);
            }
        }
        __syncthreads();
    }

#pragma unroll
    for (int t = 0; t < 4; ++t) {
        const int col = nblk + t * 16 + lo16;
        const float bv = bias[col];
#pragma unroll
        for (int r = 0; r < 8; ++r) {
            const int row = mblk + wave * 16 + r + 8 * hi;
            float v = acc[t][r] + bv;
            if (RELU) v = fmaxf(v, 0.f);
            if (HAS_RES) {
                const int rr = RBROAD ? ((row >> 12) * 512 + (row & 511)) : row;
                v += resid[((size_t)rr << 8) + col];
            }
            if (OUTF) outF[((size_t)row << 8) + col] = v;
            if (OUTB) outB[((size_t)row << 8) + col] = (__bf16)v;
            if (OUTT) outT[(size_t)col * (size_t)M + row] = (__bf16)v;
        }
    }
}

// ---------------------------------------------------------------------------
// Flash attention: grid = b*n*h*(512/128); block = 8 waves, wave = 16 q rows.
// Q,K: [4096][256] bf16 (col = h*64+d). VT: [256][4096] bf16. CTX: [32768][256].
// ---------------------------------------------------------------------------
__global__ __launch_bounds__(256) void attn_k(
    const __bf16* __restrict__ Q, const __bf16* __restrict__ Kk,
    const __bf16* __restrict__ VT, __bf16* __restrict__ CTX)
{
    __shared__ __bf16 pbuf[8][16][32];   // per-wave P tile (C-frag -> A-frag)

    const int wave = threadIdx.x >> 5, lane = threadIdx.x & 31;
    const int lo16 = lane & 15, hi = lane >> 4;
    const int blk = blockIdx.x;
    const int qt = blk & 3;
    const int h  = (blk >> 2) & 3;
    const int n  = (blk >> 4) & 7;
    const int b  = blk >> 7;
    const int q0 = qt * 128 + wave * 16;

    const __bf16* qp = Q + ((size_t)(b * 512 + q0 + lo16) << 8) + h * 64;
    const v16bf qA0 = load_a16(qp + hi * 16);
    const v16bf qA1 = load_a16(qp + 32 + hi * 16);

    v8f ctx[4];
#pragma unroll
    for (int t = 0; t < 4; ++t) ctx[t] = v8f{};
    float mrun[8], lrun[8];
#pragma unroll
    for (int r = 0; r < 8; ++r) { mrun[r] = -1e30f; lrun[r] = 0.f; }
    const float scale = 0.125f;  // 1/sqrt(64)

    for (int j0 = 0; j0 < 512; j0 += 32) {
        v8f s[2];
#pragma unroll
        for (int jt = 0; jt < 2; ++jt) {
            const __bf16* kp =
                Kk + ((size_t)(n * 512 + j0 + jt * 16 + lo16) << 8) + h * 64;
            v8f z = v8f{};
            z = wmma_bf16(qA0, load_b16(kp + hi * 16), z);
            z = wmma_bf16(qA1, load_b16(kp + 32 + hi * 16), z);
            s[jt] = z;
        }

        float fac[8];
#pragma unroll
        for (int r = 0; r < 8; ++r) {
            const float s0 = s[0][r] * scale;
            const float s1 = s[1][r] * scale;
            float mx = fmaxf(s0, s1);
#pragma unroll
            for (int m = 1; m < 16; m <<= 1) mx = fmaxf(mx, __shfl_xor(mx, m, 32));
            const float mnew = fmaxf(mrun[r], mx);
            fac[r] = __expf(mrun[r] - mnew);
            mrun[r] = mnew;
            const float p0 = __expf(s0 - mnew);
            const float p1 = __expf(s1 - mnew);
            float rs = p0 + p1;
#pragma unroll
            for (int m = 1; m < 16; m <<= 1) rs += __shfl_xor(rs, m, 32);
            lrun[r] = lrun[r] * fac[r] + rs;
            pbuf[wave][r + 8 * hi][lo16]      = (__bf16)p0;
            pbuf[wave][r + 8 * hi][16 + lo16] = (__bf16)p1;
        }
#pragma unroll
        for (int t = 0; t < 4; ++t)
#pragma unroll
            for (int r = 0; r < 8; ++r) ctx[t][r] *= fac[r];

        // same-wave DS ordering: no barrier needed for private P tile
        const v16bf pA = load_a16(&pbuf[wave][lo16][hi * 16]);
#pragma unroll
        for (int t = 0; t < 4; ++t) {
            const __bf16* vp = VT + ((size_t)(h * 64 + t * 16 + lo16) << 12) +
                               n * 512 + j0 + hi * 16;
            ctx[t] = wmma_bf16(pA, load_b16(vp), ctx[t]);
        }
    }

#pragma unroll
    for (int t = 0; t < 4; ++t)
#pragma unroll
        for (int r = 0; r < 8; ++r) {
            const float v = ctx[t][r] / lrun[r];
            const int row = (b * 8 + n) * 512 + q0 + r + 8 * hi;
            const int col = h * 64 + t * 16 + lo16;
            CTX[((size_t)row << 8) + col] = (__bf16)v;
        }
}

// ---------------------------------------------------------------------------
extern "C" void kernel_launch(void* const* d_in, const int* in_sizes, int n_in,
                              void* d_out, int out_size, void* d_ws, size_t ws_size,
                              hipStream_t stream)
{
    const float* code = (const float*)d_in[0];
    const float* tex  = (const float*)d_in[1];
    const float* Wq = (const float*)d_in[2];  const float* bq = (const float*)d_in[3];
    const float* Wk = (const float*)d_in[4];  const float* bk = (const float*)d_in[5];
    const float* Wv = (const float*)d_in[6];  const float* bv = (const float*)d_in[7];
    const float* Wo = (const float*)d_in[8];  const float* bo = (const float*)d_in[9];
    const float* ln1g = (const float*)d_in[10]; const float* ln1b = (const float*)d_in[11];
    const float* ln2g = (const float*)d_in[12]; const float* ln2b = (const float*)d_in[13];
    const float* ffg  = (const float*)d_in[14]; const float* ffb  = (const float*)d_in[15];
    const float* W1 = (const float*)d_in[16]; const float* b1 = (const float*)d_in[17];
    const float* W2 = (const float*)d_in[18]; const float* b2 = (const float*)d_in[19];
    float* out = (float*)d_out;

    // workspace layout (peak ~65MB) with lifetime-based reuse
    char* ws = (char*)d_ws;
    const size_t MB = 1024ull * 1024ull;
    float*  xf    = (float*)(ws);                 // [32768][256] f32  (0..32MB)
    __bf16* ctx   = (__bf16*)(ws + 32 * MB);      // [32768][256] bf16
    __bf16* xn    = (__bf16*)(ws + 32 * MB);      // reuses ctx after it dies
    __bf16* code2 = (__bf16*)(ws + 48 * MB);      // [4096][256]
    __bf16* tex2  = (__bf16*)(ws + 50 * MB);
    __bf16* qb    = (__bf16*)(ws + 52 * MB);
    __bf16* kb    = (__bf16*)(ws + 54 * MB);
    __bf16* vT    = (__bf16*)(ws + 56 * MB);      // [256][4096]
    __bf16* h1    = (__bf16*)(ws + 48 * MB);      // reuses code2..vT after attn
    __bf16* wbuf  = (__bf16*)(ws + 64 * MB);      // 6 x 64K bf16 weights
    __bf16* Wqb = wbuf;             __bf16* Wkb = wbuf + 1 * 65536;
    __bf16* Wvb = wbuf + 2 * 65536; __bf16* Wob = wbuf + 3 * 65536;
    __bf16* W1b = wbuf + 4 * 65536; __bf16* W2b = wbuf + 5 * 65536;

    // 0: weights f32 -> bf16 (once per launch; tiny)
    cvt_bf16_k<<<32, 256, 0, stream>>>(Wq, Wqb);
    cvt_bf16_k<<<32, 256, 0, stream>>>(Wk, Wkb);
    cvt_bf16_k<<<32, 256, 0, stream>>>(Wv, Wvb);
    cvt_bf16_k<<<32, 256, 0, stream>>>(Wo, Wob);
    cvt_bf16_k<<<32, 256, 0, stream>>>(W1, W1b);
    cvt_bf16_k<<<32, 256, 0, stream>>>(W2, W2b);

    // 1-2: LayerNorms -> bf16
    ln_bf16_k<<<512, 256, 0, stream>>>(code, ln1g, ln1b, code2, 4096);
    ln_bf16_k<<<512, 256, 0, stream>>>(tex,  ln2g, ln2b, tex2,  4096);

    // 3-5: Q/K/V projections (V stored transposed for attention B-operand)
    dim3 g1(4096 / 128, 256 / 64);
    gemm_bf16_k<false,false,false,true,false,false>
        <<<g1, 256, 0, stream>>>(tex2,  Wqb, bq, nullptr, qb, nullptr, nullptr, 4096);
    gemm_bf16_k<false,false,false,true,false,false>
        <<<g1, 256, 0, stream>>>(code2, Wkb, bk, nullptr, kb, nullptr, nullptr, 4096);
    gemm_bf16_k<false,false,false,false,true,false>
        <<<g1, 256, 0, stream>>>(code2, Wvb, bv, nullptr, nullptr, vT, nullptr, 4096);

    // 6: flash attention -> ctx [32768][256]
    attn_k<<<1024, 256, 0, stream>>>(qb, kb, vT, ctx);

    // 7: feat = ctx @ Wo^T + bo;  x = texture(broadcast rows) + feat  (f32)
    dim3 g2(32768 / 128, 256 / 64);
    gemm_bf16_k<false,true,true,false,false,true>
        <<<g2, 256, 0, stream>>>(ctx, Wob, bo, tex, nullptr, nullptr, xf, 32768);

    // 8: LN(x) -> xn
    ln_bf16_k<<<4096, 256, 0, stream>>>(xf, ffg, ffb, xn, 32768);

    // 9: h1 = relu(xn @ W1^T + b1)
    gemm_bf16_k<true,false,false,true,false,false>
        <<<g2, 256, 0, stream>>>(xn, W1b, b1, nullptr, h1, nullptr, nullptr, 32768);

    // 10: out = x + (h1 @ W2^T + b2)
    gemm_bf16_k<false,false,true,false,false,true>
        <<<g2, 256, 0, stream>>>(h1, W2b, b2, xf, nullptr, nullptr, out, 32768);
}
